// HierarchicalNAM_36129264894763
// MI455X (gfx1250) — matrix operand order repaired and checked by reading, not compiled, not run on hardware
//
#include <hip/hip_runtime.h>

typedef __attribute__((ext_vector_type(16))) _Float16 v16h;
typedef __attribute__((ext_vector_type(8)))  float    v8f;

#define BATCH 2048
#define NF    64
#define NC    10
#define NS    5
#define NG    15        // NC + NS
#define HID1  32
#define HID2  16
#define W2ELEMS (NG * NF * HID1 * HID2)          // 491520
#define W1ELEMS (NG * NF * HID1)                 // 30720
#define W2BYTES (W2ELEMS * 2)                    // 983040
#define W1BYTES (W1ELEMS * 2)                    // 61440
// ws offsets (bytes), all 256-aligned
#define OFF_W2  0
#define OFF_W1  (W2BYTES)                        // 983040
#define OFF_B1  (OFF_W1 + W1BYTES)               // 1044480
#define OFF_PART (OFF_B1 + W1BYTES)              // 1105920

// ---------------------------------------------------------------------------
// Prep 1: transpose W2 (g,f,k,n) f32 -> (g*f, n, k) f16, so the WMMA
// B-operand per lane (fixed n, 16 consecutive k) is 32 contiguous bytes.
// ---------------------------------------------------------------------------
__global__ __launch_bounds__(256) void w2_transpose_kernel(
    const float* __restrict__ cW2, const float* __restrict__ sW2,
    _Float16* __restrict__ wsW2)
{
    const int t = blockIdx.x * 256 + threadIdx.x;
    if (t >= W2ELEMS) return;
    const int gf = t >> 9;          // 0..959
    const int r  = t & 511;
    const int n  = r >> 5;          // 0..15
    const int k  = r & 31;          // 0..31
    const float* src = (gf < NC * NF) ? (cW2 + gf * (HID1 * HID2))
                                      : (sW2 + (gf - NC * NF) * (HID1 * HID2));
    wsW2[t] = (_Float16)src[k * HID2 + n];
}

// ---------------------------------------------------------------------------
// Prep 2: W1 and b1 -> f16 in the WMMA A-element order, 16 values per
// (gf, lane-half): element e <-> k = half*16 + e.
// ---------------------------------------------------------------------------
__global__ __launch_bounds__(256) void w1_pack_kernel(
    const float* __restrict__ cW1, const float* __restrict__ cB1,
    const float* __restrict__ sW1, const float* __restrict__ sB1,
    _Float16* __restrict__ wsW1, _Float16* __restrict__ wsB1)
{
    const int t = blockIdx.x * 256 + threadIdx.x;
    if (t >= W1ELEMS) return;
    const int gf   = t >> 5;        // 0..959
    const int r    = t & 31;
    const int half = r >> 4;
    const int e    = r & 15;
    const int k    = half * 16 + e;
    const bool cat = (gf < NC * NF);
    const int  lof = cat ? gf * HID1 : (gf - NC * NF) * HID1;
    wsW1[t] = (_Float16)(cat ? cW1[lof + k] : sW1[lof + k]);
    wsB1[t] = (_Float16)(cat ? cB1[lof + k] : sB1[lof + k]);
}

// ---------------------------------------------------------------------------
// Main: one block = one 32-sample tile x one group; 8 waves x 8 features each.
// Per feature: packed-f16 layer1 (pk_fma/pk_max) -> two A operands, shared
// f16 W2 B-operand, b2 in C, two v_wmma_f32_16x16x32_f16, relu via a single
// inline v_med3_num_f32 (clamp to [0,inf)), per-lane accumulate; block
// reduction once through LDS.
// ---------------------------------------------------------------------------
__global__ __launch_bounds__(256) void nam_group_kernel(
    const float* __restrict__ x,
    const float* __restrict__ cB2, const float* __restrict__ cW3,
    const float* __restrict__ cB3, const float* __restrict__ cbias,
    const float* __restrict__ sB2, const float* __restrict__ sW3,
    const float* __restrict__ sB3, const float* __restrict__ sbias,
    const _Float16* __restrict__ wsW2,
    const _Float16* __restrict__ wsW1,
    const _Float16* __restrict__ wsB1,
    float* __restrict__ wsPart)
{
    const int lane = threadIdx.x & 31;
    const int wave = threadIdx.x >> 5;
    const int half = lane >> 4;     // 16-lane group
    const int n    = lane & 15;     // N column (H2 unit) / sample row for loads
    const int tile = blockIdx.x;    // 32-sample tile
    const int g    = blockIdx.y;    // 0..14

    const float *B2, *W3, *B3;
    float wgt, gbias;
    int gl;
    if (g < NC) {
        gl = g;
        B2 = cB2; W3 = cW3; B3 = cB3;
        wgt = 0.7f / (float)NC; gbias = cbias[gl];
    } else {
        gl = g - NC;
        B2 = sB2; W3 = sW3; B3 = sB3;
        wgt = 0.3f / (float)NS; gbias = sbias[gl];
    }

    const int sampleBase = tile * 32;
    const int fBase  = wave * 8;
    const int gfBase = gl * NF + fBase;

    // Hoisted: this lane's 8 feature values for BOTH 16-sample M-tiles
    const float4* xr0 = (const float4*)(x + (sampleBase + n) * NF + fBase);
    const float4* xr1 = (const float4*)(x + (sampleBase + 16 + n) * NF + fBase);
    const float4 xa0 = xr0[0], xa1 = xr0[1];
    const float4 xb0 = xr1[0], xb1 = xr1[1];
    const float xv0[8] = { xa0.x, xa0.y, xa0.z, xa0.w, xa1.x, xa1.y, xa1.z, xa1.w };
    const float xv1[8] = { xb0.x, xb0.y, xb0.z, xb0.w, xb1.x, xb1.y, xb1.z, xb1.w };

    // Hoisted: sum of b3 over this wave's 8 features (lane-uniform)
    const float4* b3r = (const float4*)(B3 + gfBase);
    const float4 b3a = b3r[0], b3b = b3r[1];
    const float sumB3 = (b3a.x + b3a.y) + (b3a.z + b3a.w)
                      + (b3b.x + b3b.y) + (b3b.z + b3b.w);

    float acc0[8], acc1[8];
    #pragma unroll
    for (int r = 0; r < 8; ++r) { acc0[r] = 0.0f; acc1[r] = 0.0f; }

    const float POSINF = __builtin_inff();

    for (int i = 0; i < 8; ++i) {
        const int gf = gfBase + i;

        // ---- Shared f16 weight loads (A-ordered W1/b1, B-ordered W2) ----
        const v16h w1h = *(const v16h*)(wsW1 + gf * HID1 + half * 16);
        const v16h b1h = *(const v16h*)(wsB1 + gf * HID1 + half * 16);
        const v16h Bm  = *(const v16h*)(wsW2 + gf * (HID1 * HID2)
                                        + n * HID1 + half * 16);

        // C operand = b2[n] broadcast across the 8 M rows (shared)
        const float b2n = B2[gf * HID2 + n];
        v8f c;
        #pragma unroll
        for (int r = 0; r < 8; ++r) c[r] = b2n;

        const float w3n = W3[gf * HID2 + n];

        // ---- Layer 1 in packed f16: A = max(x*W1 + b1, 0) ----
        const _Float16 xh0 = (_Float16)xv0[i];
        const _Float16 xh1 = (_Float16)xv1[i];
        v16h xs0, xs1;
        #pragma unroll
        for (int e = 0; e < 16; ++e) { xs0[e] = xh0; xs1[e] = xh1; }
        const v16h zz = {};
        v16h A0 = __builtin_elementwise_max((v16h)(xs0 * w1h + b1h), zz);
        v16h A1 = __builtin_elementwise_max((v16h)(xs1 * w1h + b1h), zz);

        // D = h1 @ W2 + b2 (f32 accumulate), two M-tiles
        const v8f d0 = __builtin_amdgcn_wmma_f32_16x16x32_f16(
            false, A0, false, Bm, (short)0, c, false, false);
        const v8f d1 = __builtin_amdgcn_wmma_f32_16x16x32_f16(
            false, A1, false, Bm, (short)0, c, false, false);

        // ---- Layer 3: relu as a single v_med3_num_f32 (unfoldable), then
        //      per-lane partial accumulate; n-reduction deferred ----
        #pragma unroll
        for (int r = 0; r < 8; ++r) {
            float t0, t1;
            asm("v_med3_num_f32 %0, %1, 0, %2" : "=v"(t0) : "v"(d0[r]), "v"(POSINF));
            asm("v_med3_num_f32 %0, %1, 0, %2" : "=v"(t1) : "v"(d1[r]), "v"(POSINF));
            acc0[r] = fmaf(t0, w3n, acc0[r]);
            acc1[r] = fmaf(t1, w3n, acc1[r]);
        }
    }

    // ---- Block-wide reduction through LDS ----
    // acc0[r] -> sample m = half*8 + r ; acc1[r] -> sample m = 16 + half*8 + r
    __shared__ __align__(16) float red2[8][16][32];   // [wave][n][m], 16 KB
    __shared__ float red3[8];                          // per-wave sum of b3
    __shared__ float part[8][32];                      // per-wave per-sample

    float* dst = &red2[wave][n][half * 8];
    *(float4*)(dst)      = make_float4(acc0[0], acc0[1], acc0[2], acc0[3]);
    *(float4*)(dst + 4)  = make_float4(acc0[4], acc0[5], acc0[6], acc0[7]);
    *(float4*)(dst + 16) = make_float4(acc1[0], acc1[1], acc1[2], acc1[3]);
    *(float4*)(dst + 20) = make_float4(acc1[4], acc1[5], acc1[6], acc1[7]);
    if (lane == 0) red3[wave] = sumB3;
    __syncthreads();

    const int t = threadIdx.x;
    {   // reduce over n: 8 waves x 32 samples, all 256 threads active
        const int w = t >> 5, m = t & 31;
        float s = 0.0f;
        #pragma unroll
        for (int nn = 0; nn < 16; ++nn) s += red2[w][nn][m];
        part[w][m] = s;
    }
    __syncthreads();

    if (t < 32) {   // reduce over waves, add biases
        float s = 0.0f, sb = 0.0f;
        #pragma unroll
        for (int w = 0; w < 8; ++w) { s += part[w][t]; sb += red3[w]; }
        wsPart[g * BATCH + sampleBase + t] = wgt * (s + sb + gbias);
    }
}

// out[b] = sum over 15 groups of the weighted per-group partials
__global__ __launch_bounds__(256) void nam_reduce_kernel(
    const float* __restrict__ wsPart, float* __restrict__ out)
{
    const int b = blockIdx.x * blockDim.x + threadIdx.x;
    if (b < BATCH) {
        float s = 0.0f;
        #pragma unroll
        for (int g = 0; g < NG; ++g) s += wsPart[g * BATCH + b];
        out[b] = s;
    }
}

extern "C" void kernel_launch(void* const* d_in, const int* in_sizes, int n_in,
                              void* d_out, int out_size, void* d_ws, size_t ws_size,
                              hipStream_t stream) {
    const float* x     = (const float*)d_in[0];
    const float* cW1   = (const float*)d_in[1];
    const float* cB1   = (const float*)d_in[2];
    const float* cW2   = (const float*)d_in[3];
    const float* cB2   = (const float*)d_in[4];
    const float* cW3   = (const float*)d_in[5];
    const float* cB3   = (const float*)d_in[6];
    const float* cbias = (const float*)d_in[7];
    const float* sW1   = (const float*)d_in[8];
    const float* sB1   = (const float*)d_in[9];
    const float* sW2   = (const float*)d_in[10];
    const float* sB2   = (const float*)d_in[11];
    const float* sW3   = (const float*)d_in[12];
    const float* sB3   = (const float*)d_in[13];
    const float* sbias = (const float*)d_in[14];

    _Float16* wsW2   = (_Float16*)((char*)d_ws + OFF_W2);
    _Float16* wsW1   = (_Float16*)((char*)d_ws + OFF_W1);
    _Float16* wsB1   = (_Float16*)((char*)d_ws + OFF_B1);
    float*    wsPart = (float*)   ((char*)d_ws + OFF_PART);
    float*    out    = (float*)d_out;

    w2_transpose_kernel<<<(W2ELEMS + 255) / 256, 256, 0, stream>>>(cW2, sW2, wsW2);
    w1_pack_kernel<<<(W1ELEMS + 255) / 256, 256, 0, stream>>>(
        cW1, cB1, sW1, sB1, wsW1, wsB1);

    dim3 grid(BATCH / 32, NG);   // 64 tiles x 15 groups
    nam_group_kernel<<<grid, 256, 0, stream>>>(
        x, cB2, cW3, cB3, cbias, sB2, sW3, sB3, sbias,
        wsW2, wsW1, wsB1, wsPart);

    nam_reduce_kernel<<<(BATCH + 255) / 256, 256, 0, stream>>>(wsPart, out);
}
